// _CRLI_78864189489887
// MI455X (gfx1250) — matrix operand order repaired and checked by reading, not compiled, not run on hardware
//
#include <hip/hip_runtime.h>

typedef unsigned short u16;
typedef __attribute__((ext_vector_type(16))) __bf16 bfrag;
typedef __attribute__((ext_vector_type(8)))  float  v8f;

#define BZ     4096
#define NS     128
#define DIN    32
#define DH     128
#define DG     384
#define DO     32
#define NTH    256
#define SMTH   1024
#define NSLICE 16

static __device__ __forceinline__ u16 f2bf_raw(float f) {
  unsigned int u = __builtin_bit_cast(unsigned int, f);
  u += 0x7FFFu + ((u >> 16) & 1u);          // round-to-nearest-even
  return (u16)(u >> 16);
}
static __device__ __forceinline__ void vzero(v8f& a) {
#pragma unroll
  for (int i = 0; i < 8; ++i) a[i] = 0.0f;
}
static __device__ __forceinline__ float sigmoidf_(float x) {
  return 1.0f / (1.0f + __expf(-x));
}
static __device__ __forceinline__ float tanh_fast(float x) {
  const float ax = fabsf(x);
  const float r = 1.0f - 2.0f / (__expf(2.0f * ax) + 1.0f);
  return __builtin_copysignf(r, x);
}

// Assemble a 16 x bf16 fragment from two contiguous 16-byte runs.
union fragcast { uint4 u[2]; bfrag f; };
static __device__ __forceinline__ bfrag load_frag(const u16* p0, const u16* p1) {
  fragcast c;
  c.u[0] = *(const uint4*)p0;
  c.u[1] = *(const uint4*)p1;
  return c.f;
}

// ---------------------------------------------------------------------------
// C(32 x N) += A_lds(32 x K bf16, row-major) @ B (fragment-ready global bf16)
// Fragment-ready B layout: wf[((kc*(N/16)+tj)*16+col)*32 + k] = W[kc*32+k][tj*16+col]
// Wave w owns column tiles tj = w, w+8, w+16; acc index always compile-time.
// ---------------------------------------------------------------------------
template<int K, int N>
static __device__ __forceinline__ void mm_tiles(
    const u16* __restrict__ sA, const u16* __restrict__ gBf,
    v8f* acc, int wave, int lane)
{
  constexpr int TJ = N / 16;
  const int am   = lane & 15;
  const int koff = (lane >> 4) << 3;
  const int kb   = (lane >> 4) << 4;
  const int bn   = lane & 15;
  if constexpr (N == 32) {
    if (wave >= 4) return;
    const int ti = wave >> 1, tj = wave & 1;
#pragma unroll
    for (int kc = 0; kc < K / 32; ++kc) {
      const u16* pa = sA + (ti * 16 + am) * K + kc * 32;
      const u16* pb = gBf + ((size_t)(kc * TJ + tj) * 16 + bn) * 32 + koff;
      acc[0] = __builtin_amdgcn_wmma_f32_16x16x32_bf16(
          false, load_frag(pa + koff, pa + 16 + koff),
          false, load_frag(pb, pb + 16), (short)0, acc[0], false, false);
    }
  } else {
    constexpr int JW = TJ / 8;   // column tiles per wave (1 for N=128, 3 for N=384)
#pragma unroll
    for (int kc = 0; kc < K / 32; ++kc) {
      const u16* pa0 = sA + am * K + kc * 32;
      const u16* pa1 = pa0 + 16 * K;
      const bfrag A0 = load_frag(pa0 + koff, pa0 + 16 + koff);
      const bfrag A1 = load_frag(pa1 + koff, pa1 + 16 + koff);
#pragma unroll
      for (int j = 0; j < JW; ++j) {
        const int tj = wave + 8 * j;
        const u16* pb = gBf + ((size_t)(kc * TJ + tj) * 16 + bn) * 32 + koff;
        const bfrag B = load_frag(pb, pb + 16);
        acc[2 * j + 0] = __builtin_amdgcn_wmma_f32_16x16x32_bf16(
            false, A0, false, B, (short)0, acc[2 * j + 0], false, false);
        acc[2 * j + 1] = __builtin_amdgcn_wmma_f32_16x16x32_bf16(
            false, A1, false, B, (short)0, acc[2 * j + 1], false, false);
      }
    }
  }
}

// Scores tail: part[wg] = h2_tile^T @ enc_tile (K = 32 local batch rows).
static __device__ __forceinline__ void scores_tail(
    const u16* __restrict__ s_hT, const u16* __restrict__ s_eT,
    float* __restrict__ part, int wg, int wave, int lane)
{
  const int am = lane & 15, koff = (lane >> 4) << 3;
  const int kb = (lane >> 4) << 4, bn = lane & 15;
  v8f sacc[8];
#pragma unroll
  for (int q = 0; q < 8; ++q) vzero(sacc[q]);
  const u16* pb = s_eT + (wave * 16 + bn) * 32 + koff;
  const bfrag B = load_frag(pb, pb + 16);
#pragma unroll
  for (int q = 0; q < 8; ++q) {
    const u16* pa = s_hT + (q * 16 + am) * 32;
    sacc[q] = __builtin_amdgcn_wmma_f32_16x16x32_bf16(
        false, load_frag(pa + koff, pa + 16 + koff),
        false, B, (short)0, sacc[q], false, false);
  }
  const int m0 = (lane >> 4) << 3, n = lane & 15;
#pragma unroll
  for (int q = 0; q < 8; ++q)
#pragma unroll
    for (int v = 0; v < 8; ++v)
      part[((size_t)wg * DH + q * 16 + m0 + v) * DH + wave * 16 + n] = sacc[q][v];
}

// Pack a K x N weight into fragment-ready bf16 layout.
template<int K, int N, typename F>
static __device__ __forceinline__ void frag_pack(int t0, int stride, u16* dst, F src) {
  for (int i = t0; i < K * N; i += stride) {
    const int k    = i & 31;
    const int col  = (i >> 5) & 15;
    const int tjkc = i >> 9;
    const int tj   = tjkc % (N / 16);
    const int kc   = tjkc / (N / 16);
    dst[i] = f2bf_raw(src(kc * 32 + k, tj * 16 + col));
  }
}

// ---- weight conversion / swizzle to fragment-ready bf16 (once) --------------
__global__ void convert_kernel(
    const float* __restrict__ fcn, const float* __restrict__ att1_w,
    const float* __restrict__ att2_w, const float* __restrict__ g1_wih,
    const float* __restrict__ g1_whh, const float* __restrict__ g2_wih,
    const float* __restrict__ g2_whh, const float* __restrict__ out_w,
    u16* fcn_bf, u16* wc1T, u16* wh1T, u16* wih1T, u16* whh1T,
    u16* wc2T, u16* wh2T, u16* wih2T, u16* whh2T, u16* outT)
{
  const int stride = gridDim.x * blockDim.x;
  const int t0 = blockIdx.x * blockDim.x + threadIdx.x;
  for (int i = t0; i < BZ * DIN; i += stride) fcn_bf[i] = f2bf_raw(fcn[i]);
  frag_pack<DIN, DIN>(t0, stride, wc1T,
      [&](int k, int n) { return att1_w[n * (2 * DIN) + k]; });
  frag_pack<DIN, DIN>(t0, stride, wh1T,
      [&](int k, int n) { return att1_w[n * (2 * DIN) + DIN + k]; });
  frag_pack<DIN, DG>(t0, stride, wih1T,
      [&](int k, int n) { return g1_wih[n * DIN + k]; });
  frag_pack<DH, DG>(t0, stride, whh1T,
      [&](int k, int n) { return g1_whh[n * DH + k]; });
  frag_pack<DH, DH>(t0, stride, wc2T,
      [&](int k, int n) { return att2_w[n * (2 * DH) + k]; });
  frag_pack<DH, DH>(t0, stride, wh2T,
      [&](int k, int n) { return att2_w[n * (2 * DH) + DH + k]; });
  frag_pack<DH, DG>(t0, stride, wih2T,
      [&](int k, int n) { return g2_wih[n * DH + k]; });
  frag_pack<DH, DG>(t0, stride, whh2T,
      [&](int k, int n) { return g2_whh[n * DH + k]; });
  frag_pack<DH, DO>(t0, stride, outT,
      [&](int k, int n) { return out_w[n * DH + k]; });
}

// ---- initial scores: part[slice] = fcn^T @ enc_0 (split-K over batch) -------
__global__ void scores0_kernel(
    const u16* __restrict__ hsrc,   // (BZ, DIN) raw bf16
    const float* __restrict__ enc,  // (BZ, NS, DH) f32
    float* __restrict__ part)       // (NSLICE, DIN, DH) f32
{
  constexpr int SB = BZ / NSLICE;
  __shared__ u16 s_hT[DIN * 32];
  __shared__ u16 s_eT[DH * 32];
  const int tid = threadIdx.x, wave = tid >> 5, lane = tid & 31;
  const int b0 = blockIdx.x * SB;
  const int am = lane & 15, koff = (lane >> 4) << 3;
  const int kb = (lane >> 4) << 4, bn = lane & 15;
  v8f acc[2]; vzero(acc[0]); vzero(acc[1]);

  for (int c = 0; c < SB; c += 32) {
    const int bb = b0 + c;
    __syncthreads();
    for (int i = tid; i < 32 * DIN / 8; i += NTH) {
      const int gi = i * 8, r = gi / DIN, cc = gi % DIN;
      uint4 raw = *(const uint4*)&hsrc[(size_t)bb * DIN + gi];
      const u16* s = (const u16*)&raw;
#pragma unroll
      for (int e = 0; e < 8; ++e) s_hT[(cc + e) * 32 + r] = s[e];
    }
    for (int i = tid; i < 32 * DH / 4; i += NTH) {
      const int gi = i * 4, r = gi / DH, cc = gi % DH;
      const float4 f = *(const float4*)&enc[((size_t)(bb + r) * NS + 0) * DH + cc];
      const u16 v4[4] = { f2bf_raw(f.x), f2bf_raw(f.y), f2bf_raw(f.z), f2bf_raw(f.w) };
#pragma unroll
      for (int e = 0; e < 4; ++e) s_eT[(cc + e) * 32 + r] = v4[e];
    }
    __syncthreads();
    const u16* pb = s_eT + (wave * 16 + bn) * 32 + koff;
    const bfrag B = load_frag(pb, pb + 16);
#pragma unroll
    for (int q = 0; q < 2; ++q) {
      const u16* pa = s_hT + (q * 16 + am) * 32;
      acc[q] = __builtin_amdgcn_wmma_f32_16x16x32_bf16(
          false, load_frag(pa + koff, pa + 16 + koff),
          false, B, (short)0, acc[q], false, false);
    }
  }
  const int m0 = (lane >> 4) << 3, n = lane & 15;
#pragma unroll
  for (int q = 0; q < 2; ++q)
#pragma unroll
    for (int v = 0; v < 8; ++v)
      part[((size_t)blockIdx.x * DIN + q * 16 + m0 + v) * DH + wave * 16 + n] = acc[q][v];
}

// ---- softmax: sum NP partials, row-softmax, emit fragment-ready wT ----------
// Grid = D/16 WGs x 1024 threads; WG b handles rows [16b,16b+16) == tile tj=b.
template<int D, int NP>
__global__ void softmax_kernel(const float* __restrict__ part, u16* __restrict__ wT)
{
  __shared__ float S[16 * DH];
  __shared__ float red[SMTH];
  __shared__ float rowv[16];
  const int tid = threadIdx.x;
  const int r0 = blockIdx.x * 16;
  {   // partial sum: one float2 per thread (1024 * 2 == 16*128)
    const int rr = (tid * 2) / DH, e = (tid * 2) % DH;
    float2 s; s.x = 0.f; s.y = 0.f;
    for (int p = 0; p < NP; ++p) {
      const float2 v = *(const float2*)&part[((size_t)p * D + r0 + rr) * DH + e];
      s.x += v.x; s.y += v.y;
    }
    *(float2*)&S[rr * DH + e] = s;
  }
  __syncthreads();
  const int row = tid >> 6, c2 = (tid & 63) * 2;   // 64 lanes per row
  float lmax = fmaxf(S[row * DH + c2], S[row * DH + c2 + 1]);
  red[tid] = lmax;
  __syncthreads();
  if ((tid & 63) == 0) {
    float m = red[tid];
    for (int s2 = 1; s2 < 64; ++s2) m = fmaxf(m, red[tid + s2]);
    rowv[row] = m;
  }
  __syncthreads();
  const float mx = rowv[row];
  const float e0 = __expf(S[row * DH + c2] - mx);
  const float e1 = __expf(S[row * DH + c2 + 1] - mx);
  S[row * DH + c2] = e0;
  S[row * DH + c2 + 1] = e1;
  red[tid] = e0 + e1;
  __syncthreads();
  if ((tid & 63) == 0) {
    float sv = 0.0f;
    for (int s2 = 0; s2 < 64; ++s2) sv += red[tid + s2];
    rowv[row] = 1.0f / sv;
  }
  __syncthreads();
  const float inv = rowv[row];
  S[row * DH + c2] *= inv;
  S[row * DH + c2 + 1] *= inv;
  __syncthreads();
  constexpr int TJ = D / 16;
  for (int i = tid; i < 16 * DH; i += SMTH) {      // pack this WG's tj slice
    const int k = i & 31, col = (i >> 5) & 15, kc = i >> 9;
    wT[((size_t)(kc * TJ + blockIdx.x) * 16 + col) * 32 + k] =
        f2bf_raw(S[col * DH + kc * 32 + k]);
  }
}

// ---- fused step: context -> attention -> GRU -> h -> out proj -> next scores
__global__ void step_kernel(
    const float* __restrict__ enc, u16* __restrict__ enc_bf,
    float* __restrict__ h_f32, u16* __restrict__ h_bf,
    const u16* __restrict__ wsmT,
    const u16* __restrict__ wcT, const u16* __restrict__ whT,
    const u16* __restrict__ wihT, const u16* __restrict__ whhT,
    const u16* __restrict__ outT,
    const float* __restrict__ att_b, const float* __restrict__ bih,
    const float* __restrict__ bhh, const float* __restrict__ out_b,
    float* __restrict__ out, float* __restrict__ part, int t)
{
  extern __shared__ char smem[];
  u16* s_e  = (u16*)smem;                 // 32 x 128   enc_t row-major
  u16* s_h  = s_e + 32 * DH;              // 32 x 128   h row-major
  u16* s_c  = s_h + 32 * DH;              // 32 x 128   context
  u16* s_a  = s_c + 32 * DH;              // 32 x 128   attention act
  u16* s_hT = s_a + 32 * DH;              // 128 x 32   h2 transposed
  u16* s_eT = s_hT + DH * 32;             // 128 x 32   enc_{t+1} transposed
  float* s_g = (float*)(s_eT + DH * 32);  // 32 x 384
  float* s_n = s_g + 32 * DG;             // 32 x 128

  const int tid = threadIdx.x, wave = tid >> 5, lane = tid & 31;
  const int b0 = blockIdx.x * 32;
  const int m0 = (lane >> 4) << 3, n = lane & 15;

  __builtin_prefetch(wihT + ((size_t)tid << 5), 0, 0);
  __builtin_prefetch(whhT + ((size_t)tid << 5), 0, 0);

  // enc_t (bf16, prepared by previous step) and h: plain vector copies
  for (int i = tid; i < 32 * DH / 8; i += NTH) {
    ((uint4*)s_h)[i] = ((const uint4*)(h_bf + (size_t)b0 * DH))[i];
    ((uint4*)s_e)[i] = ((const uint4*)(enc_bf + (size_t)b0 * DH))[i];
  }
  __syncthreads();   // all enc_bf reads done before overwrite below
  if (t + 1 < NS) {
    for (int i = tid; i < 32 * DH / 4; i += NTH) {  // enc_{t+1}: f32 -> bf16
      const int gi = i * 4, r = gi / DH, cc = gi % DH;
      const float4 f = *(const float4*)&enc[((size_t)(b0 + r) * NS + t + 1) * DH + cc];
      const u16 v4[4] = { f2bf_raw(f.x), f2bf_raw(f.y), f2bf_raw(f.z), f2bf_raw(f.w) };
#pragma unroll
      for (int e = 0; e < 4; ++e) s_eT[(cc + e) * 32 + r] = v4[e];     // transposed
      *(uint2*)&enc_bf[(size_t)(b0 + r) * DH + cc] = *(const uint2*)v4; // rolling buf
    }
  }
  __syncthreads();

  // 1) context = enc @ wsm^T
  v8f acc2[2]; vzero(acc2[0]); vzero(acc2[1]);
  mm_tiles<DH, DH>(s_e, wsmT, acc2, wave, lane);
#pragma unroll
  for (int q = 0; q < 2; ++q)
#pragma unroll
    for (int v = 0; v < 8; ++v)
      s_c[(q * 16 + m0 + v) * DH + wave * 16 + n] = f2bf_raw(acc2[q][v]);
  __syncthreads();

  // 2) a = tanh(ctx @ wc + h @ wh + att_b)
  vzero(acc2[0]); vzero(acc2[1]);
  mm_tiles<DH, DH>(s_c, wcT, acc2, wave, lane);
  mm_tiles<DH, DH>(s_h, whT, acc2, wave, lane);
#pragma unroll
  for (int q = 0; q < 2; ++q) {
    const int col = wave * 16 + n;
#pragma unroll
    for (int v = 0; v < 8; ++v)
      s_a[(q * 16 + m0 + v) * DH + col] = f2bf_raw(tanh_fast(acc2[q][v] + att_b[col]));
  }
  __syncthreads();

  // 3) gh = h @ whh^T + bhh
  v8f acc6[6];
#pragma unroll
  for (int q = 0; q < 6; ++q) vzero(acc6[q]);
  mm_tiles<DH, DG>(s_h, whhT, acc6, wave, lane);
#pragma unroll
  for (int q = 0; q < 6; ++q) {
    const int col = (wave + 8 * (q >> 1)) * 16 + n;
#pragma unroll
    for (int v = 0; v < 8; ++v)
      s_g[((q & 1) * 16 + m0 + v) * DG + col] = acc6[q][v] + bhh[col];
  }

  // 4) gi = a @ wih^T + bih  (r/z add into s_g; i_n part to s_n)
#pragma unroll
  for (int q = 0; q < 6; ++q) vzero(acc6[q]);
  mm_tiles<DH, DG>(s_a, wihT, acc6, wave, lane);
#pragma unroll
  for (int q = 0; q < 6; ++q) {
    const int col = (wave + 8 * (q >> 1)) * 16 + n;
    const int row0 = (q & 1) * 16 + m0;
#pragma unroll
    for (int v = 0; v < 8; ++v) {
      const float val = acc6[q][v] + bih[col];
      if ((q >> 1) < 2) s_g[(row0 + v) * DG + col] += val;
      else              s_n[(row0 + v) * DH + (col - 2 * DH)] = val;
    }
  }
  __syncthreads();

  // 5) GRU gates (f32 carry for h); also write transposed h2 for scores tail
  for (int i = tid; i < 32 * DH; i += NTH) {
    const int r = i >> 7, d = i & 127;
    const float rg = sigmoidf_(s_g[r * DG + d]);
    const float z  = sigmoidf_(s_g[r * DG + DH + d]);
    const float nn = tanh_fast(s_n[i] + rg * s_g[r * DG + 2 * DH + d]);
    const float hp = h_f32[(size_t)(b0 + r) * DH + d];
    const float h2 = (1.0f - z) * nn + z * hp;
    h_f32[(size_t)(b0 + r) * DH + d] = h2;
    const u16 hb = f2bf_raw(h2);
    h_bf[(size_t)(b0 + r) * DH + d] = hb;
    s_h[i] = hb;
    s_hT[d * 32 + r] = hb;
  }
  __syncthreads();

  // 6) recon[:, t, :] = h2 @ out_w^T + out_b
  v8f acco[1]; vzero(acco[0]);
  mm_tiles<DH, DO>(s_h, outT, acco, wave, lane);
  if (wave < 4) {
    const int ti = wave >> 1, col = (wave & 1) * 16 + n;
#pragma unroll
    for (int v = 0; v < 8; ++v)
      out[((size_t)(b0 + ti * 16 + m0 + v) * NS + t) * DO + col] = acco[0][v] + out_b[col];
  }

  // 7) next-step scores partial: part[wg] = h2^T @ enc_{t+1}
  if (t + 1 < NS) scores_tail(s_hT, s_eT, part, blockIdx.x, wave, lane);
}

// ---- step 0 (D_IN-shaped attention + GRU1 with h_rep = tile(fcn,4)) ---------
__global__ void step0_kernel(
    const float* __restrict__ fcn, const u16* __restrict__ fcn_bf,
    const float* __restrict__ enc, u16* __restrict__ enc_bf,
    float* __restrict__ h_f32, u16* __restrict__ h_bf,
    const u16* __restrict__ wsm0T,
    const u16* __restrict__ wc1T, const u16* __restrict__ wh1T,
    const u16* __restrict__ wih1T, const u16* __restrict__ whh1T,
    const u16* __restrict__ outT,
    const float* __restrict__ att1_b, const float* __restrict__ bih,
    const float* __restrict__ bhh, const float* __restrict__ out_b,
    float* __restrict__ out, float* __restrict__ part)
{
  extern __shared__ char smem[];
  u16* s_e  = (u16*)smem;                 // 32 x 128  enc_0 row-major
  u16* s_h0 = s_e + 32 * DH;              // 32 x 32
  u16* s_c  = s_h0 + 32 * DIN;            // 32 x 32
  u16* s_a  = s_c + 32 * DIN;             // 32 x 32
  u16* s_hr = s_a + 32 * DIN;             // 32 x 128  h_rep, later new h
  u16* s_hT = s_hr + 32 * DH;             // 128 x 32  h2 transposed
  u16* s_eT = s_hT + DH * 32;             // 128 x 32  enc_1 transposed
  float* s_g = (float*)(s_eT + DH * 32);  // 32 x 384
  float* s_n = s_g + 32 * DG;             // 32 x 128

  const int tid = threadIdx.x, wave = tid >> 5, lane = tid & 31;
  const int b0 = blockIdx.x * 32;
  const int m0 = (lane >> 4) << 3, n = lane & 15;

  for (int i = tid; i < 32 * DH / 4; i += NTH) {      // enc_0 -> bf16 row-major
    const int gi = i * 4, r = gi / DH, cc = gi % DH;
    const float4 f = *(const float4*)&enc[((size_t)(b0 + r) * NS + 0) * DH + cc];
    const u16 v4[4] = { f2bf_raw(f.x), f2bf_raw(f.y), f2bf_raw(f.z), f2bf_raw(f.w) };
    *(uint2*)&s_e[r * DH + cc] = *(const uint2*)v4;
  }
  for (int i = tid; i < 32 * DH / 4; i += NTH) {      // enc_1 -> transposed + roll
    const int gi = i * 4, r = gi / DH, cc = gi % DH;
    const float4 f = *(const float4*)&enc[((size_t)(b0 + r) * NS + 1) * DH + cc];
    const u16 v4[4] = { f2bf_raw(f.x), f2bf_raw(f.y), f2bf_raw(f.z), f2bf_raw(f.w) };
#pragma unroll
    for (int e = 0; e < 4; ++e) s_eT[(cc + e) * 32 + r] = v4[e];
    *(uint2*)&enc_bf[(size_t)(b0 + r) * DH + cc] = *(const uint2*)v4;
  }
  for (int i = tid; i < 32 * DIN; i += NTH)
    s_h0[i] = fcn_bf[(size_t)b0 * DIN + i];
  for (int i = tid; i < 32 * DH; i += NTH)
    s_hr[i] = fcn_bf[(size_t)(b0 + (i >> 7)) * DIN + (i & 31)];
  __syncthreads();

  // ctx0 = enc0 @ wsm0 (32x128 @ 128x32)
  v8f acc1[1]; vzero(acc1[0]);
  mm_tiles<DH, DIN>(s_e, wsm0T, acc1, wave, lane);
  if (wave < 4) {
    const int ti = wave >> 1, col = (wave & 1) * 16 + n;
#pragma unroll
    for (int v = 0; v < 8; ++v)
      s_c[(ti * 16 + m0 + v) * DIN + col] = f2bf_raw(acc1[0][v]);
  }
  __syncthreads();

  // a0 = tanh(ctx0 @ wc1 + fcn @ wh1 + att1_b)
  vzero(acc1[0]);
  mm_tiles<DIN, DIN>(s_c, wc1T, acc1, wave, lane);
  mm_tiles<DIN, DIN>(s_h0, wh1T, acc1, wave, lane);
  if (wave < 4) {
    const int ti = wave >> 1, col = (wave & 1) * 16 + n;
#pragma unroll
    for (int v = 0; v < 8; ++v)
      s_a[(ti * 16 + m0 + v) * DIN + col] = f2bf_raw(tanh_fast(acc1[0][v] + att1_b[col]));
  }
  __syncthreads();

  // gh = h_rep @ whh1^T + bhh
  v8f acc6[6];
#pragma unroll
  for (int q = 0; q < 6; ++q) vzero(acc6[q]);
  mm_tiles<DH, DG>(s_hr, whh1T, acc6, wave, lane);
#pragma unroll
  for (int q = 0; q < 6; ++q) {
    const int col = (wave + 8 * (q >> 1)) * 16 + n;
#pragma unroll
    for (int v = 0; v < 8; ++v)
      s_g[((q & 1) * 16 + m0 + v) * DG + col] = acc6[q][v] + bhh[col];
  }

  // gi = a0 @ wih1^T + bih
#pragma unroll
  for (int q = 0; q < 6; ++q) vzero(acc6[q]);
  mm_tiles<DIN, DG>(s_a, wih1T, acc6, wave, lane);
#pragma unroll
  for (int q = 0; q < 6; ++q) {
    const int col = (wave + 8 * (q >> 1)) * 16 + n;
    const int row0 = (q & 1) * 16 + m0;
#pragma unroll
    for (int v = 0; v < 8; ++v) {
      const float val = acc6[q][v] + bih[col];
      if ((q >> 1) < 2) s_g[(row0 + v) * DG + col] += val;
      else              s_n[(row0 + v) * DH + (col - 2 * DH)] = val;
    }
  }
  __syncthreads();

  for (int i = tid; i < 32 * DH; i += NTH) {
    const int r = i >> 7, d = i & 127;
    const float rg = sigmoidf_(s_g[r * DG + d]);
    const float z  = sigmoidf_(s_g[r * DG + DH + d]);
    const float nn = tanh_fast(s_n[i] + rg * s_g[r * DG + 2 * DH + d]);
    const float hp = fcn[(size_t)(b0 + r) * DIN + (d & 31)];   // h_rep in f32
    const float h2 = (1.0f - z) * nn + z * hp;
    h_f32[(size_t)(b0 + r) * DH + d] = h2;
    const u16 hb = f2bf_raw(h2);
    h_bf[(size_t)(b0 + r) * DH + d] = hb;
    s_hr[i] = hb;
    s_hT[d * 32 + r] = hb;
  }
  __syncthreads();

  // recon[:, 0, :] = h @ out_w^T + out_b
  v8f acco[1]; vzero(acco[0]);
  mm_tiles<DH, DO>(s_hr, outT, acco, wave, lane);
  if (wave < 4) {
    const int ti = wave >> 1, col = (wave & 1) * 16 + n;
#pragma unroll
    for (int v = 0; v < 8; ++v)
      out[((size_t)(b0 + ti * 16 + m0 + v) * NS + 0) * DO + col] = acco[0][v] + out_b[col];
  }

  // scores partial for t = 1
  scores_tail(s_hT, s_eT, part, blockIdx.x, wave, lane);
}

extern "C" void kernel_launch(void* const* d_in, const int* in_sizes, int n_in,
                              void* d_out, int out_size, void* d_ws, size_t ws_size,
                              hipStream_t stream) {
  (void)in_sizes; (void)n_in; (void)out_size; (void)ws_size;
  const float* fcn    = (const float*)d_in[0];
  const float* enc    = (const float*)d_in[1];
  const float* att1_w = (const float*)d_in[2];
  const float* att1_b = (const float*)d_in[3];
  const float* att2_w = (const float*)d_in[4];
  const float* att2_b = (const float*)d_in[5];
  const float* g1_wih = (const float*)d_in[6];
  const float* g1_whh = (const float*)d_in[7];
  const float* g1_bih = (const float*)d_in[8];
  const float* g1_bhh = (const float*)d_in[9];
  const float* g2_wih = (const float*)d_in[10];
  const float* g2_whh = (const float*)d_in[11];
  const float* g2_bih = (const float*)d_in[12];
  const float* g2_bhh = (const float*)d_in[13];
  const float* out_w  = (const float*)d_in[14];
  const float* out_b  = (const float*)d_in[15];
  float* out = (float*)d_out;

  char* p = (char*)d_ws;
  auto take = [&](size_t bytes) -> char* {
    char* r = p; p += (bytes + 255) & ~(size_t)255; return r;
  };
  float* h_f32 = (float*)take((size_t)BZ * DH * 4);
  u16* h_bf    = (u16*)take((size_t)BZ * DH * 2);
  u16* enc_bf  = (u16*)take((size_t)BZ * DH * 2);
  float* part  = (float*)take((size_t)(BZ / 32) * DH * DH * 4);  // 128 partials
  u16* wsmT    = (u16*)take((size_t)DH * DH * 2);
  u16* fcn_bf  = (u16*)take((size_t)BZ * DIN * 2);
  u16* wc1T    = (u16*)take(DIN * DIN * 2);
  u16* wh1T    = (u16*)take(DIN * DIN * 2);
  u16* wih1T   = (u16*)take(DIN * DG * 2);
  u16* whh1T   = (u16*)take(DH * DG * 2);
  u16* wc2T    = (u16*)take(DH * DH * 2);
  u16* wh2T    = (u16*)take(DH * DH * 2);
  u16* wih2T   = (u16*)take(DH * DG * 2);
  u16* whh2T   = (u16*)take(DH * DG * 2);
  u16* outT    = (u16*)take(DH * DO * 2);

  convert_kernel<<<256, NTH, 0, stream>>>(fcn, att1_w, att2_w, g1_wih, g1_whh,
      g2_wih, g2_whh, out_w, fcn_bf, wc1T, wh1T, wih1T, whh1T, wc2T, wh2T,
      wih2T, whh2T, outT);

  constexpr size_t SH_STEP  = (size_t)(6 * 32 * DH) * 2 + (size_t)(32 * DG + 32 * DH) * 4;
  constexpr size_t SH_STEP0 = (size_t)(4 * 32 * DH + 3 * 32 * DIN) * 2 +
                              (size_t)(32 * DG + 32 * DH) * 4;

  scores0_kernel<<<NSLICE, NTH, 0, stream>>>(fcn_bf, enc, part);
  softmax_kernel<DIN, NSLICE><<<DIN / 16, SMTH, 0, stream>>>(part, wsmT);
  step0_kernel<<<BZ / 32, NTH, SH_STEP0, stream>>>(fcn, fcn_bf, enc, enc_bf,
      h_f32, h_bf, wsmT, wc1T, wh1T, wih1T, whh1T, outT, att1_b, g1_bih,
      g1_bhh, out_b, out, part);

  for (int t = 1; t < NS; ++t) {
    softmax_kernel<DH, BZ / 32><<<DH / 16, SMTH, 0, stream>>>(part, wsmT);
    step_kernel<<<BZ / 32, NTH, SH_STEP, stream>>>(enc, enc_bf, h_f32, h_bf,
        wsmT, wc2T, wh2T, wih2T, whh2T, outT, att2_b, g2_bih, g2_bhh, out_b,
        out, part, t);
  }
}
